// LongformerSelfAttention_89945205113162
// MI455X (gfx1250) — matrix operand order repaired and checked by reading, not compiled, not run on hardware
//
#include <hip/hip_runtime.h>

// ---------------- types / helpers ----------------
typedef __attribute__((ext_vector_type(16))) __bf16 v16bf;
typedef __attribute__((ext_vector_type(8)))  __bf16 v8bf;
typedef __attribute__((ext_vector_type(8)))  float  v8f;

#define SEQ   4096
#define EMB   1024
#define NH    16
#define HD    64
#define CH    128          // chunk (= HALF_W)
#define SCALE_F 0.125f     // 64^-0.5

__device__ inline v8f vzero() { v8f z; for (int i = 0; i < 8; ++i) z[i] = 0.f; return z; }

__device__ inline v8f wmma_bf16(v16bf a, v16bf b, v8f c) {
    return __builtin_amdgcn_wmma_f32_16x16x32_bf16(false, a, false, b, (short)0, c, false, false);
}

// A-matrix 16x32 bf16 fragment: lane part p holds K = kb + 16p + [0..15] (contiguous)
__device__ inline v16bf load_a_frag(const __bf16* rowk, int lp) {
    v8bf lo = *(const v8bf*)(rowk + lp * 16);
    v8bf hi = *(const v8bf*)(rowk + lp * 16 + 8);
    v16bf a;
    for (int i = 0; i < 8; ++i) { a[i] = lo[i]; a[8 + i] = hi[i]; }
    return a;
}
// B-matrix 32x16 bf16 fragment: lane part p holds K = kb + 16p + [0..15] (contiguous)
__device__ inline v16bf load_b_frag(const __bf16* rowk, int lp) {
    return *(const v16bf*)(rowk + lp * 16);
}

// ---------------- converts ----------------
__global__ void k_cvt(const float* __restrict__ src, __bf16* __restrict__ dst, int n) {
    int i = blockIdx.x * 256 + threadIdx.x;
    if (i < n) dst[i] = (__bf16)src[i];
}

// ctx [B,H,S,D] f32 -> ctxb [token][E] bf16 (e = h*64+d)
__global__ void k_cvt_ctx(const float* __restrict__ ctx, __bf16* __restrict__ ctxb) {
    int i = blockIdx.x * 256 + threadIdx.x;       // 8,388,608 total
    int d = i & 63, s = (i >> 6) & 4095, h = (i >> 18) & 15, b = i >> 22;
    ctxb[((size_t)(b * SEQ + s)) * EMB + h * HD + d] = (__bf16)ctx[i];
}

// ---------------- QKV projection ----------------
// grid (64, 16, 3), 256 threads (8 waves). Block: 128 rows x 64 cols (one head).
__global__ void k_qkv(const __bf16* __restrict__ xb,
                      const __bf16* __restrict__ Wqb, const __bf16* __restrict__ Wkb,
                      const __bf16* __restrict__ Wvb,
                      const float* __restrict__ bq, const float* __restrict__ bk,
                      const float* __restrict__ bv,
                      __bf16* __restrict__ Q, __bf16* __restrict__ K, __bf16* __restrict__ Vt) {
    const int which = blockIdx.z;
    const __bf16* W = (which == 0) ? Wqb : (which == 1) ? Wkb : Wvb;
    const float* bias = (which == 0) ? bq : (which == 1) ? bk : bv;
    const int m0 = blockIdx.x * 128;
    const int h  = blockIdx.y;
    const int tid = threadIdx.x, w = tid >> 5, l = tid & 31, lp = l >> 4, ln = l & 15;
    const int arowi = m0 + w * 16 + ln;
    const __bf16* arow = xb + (size_t)arowi * EMB;

    v8f acc[4]; for (int t = 0; t < 4; ++t) acc[t] = vzero();
    for (int kb = 0; kb < EMB; kb += 32) {
        v16bf a = load_a_frag(arow + kb, lp);
        for (int nt = 0; nt < 4; ++nt) {
            int n = h * HD + nt * 16 + ln;
            v16bf bfr = load_b_frag(W + (size_t)n * EMB + kb, lp);
            acc[nt] = wmma_bf16(a, bfr, acc[nt]);
        }
    }
    for (int nt = 0; nt < 4; ++nt) {
        int d = nt * 16 + ln;
        float bval = bias[h * HD + d];
        for (int r = 0; r < 8; ++r) {
            int token = m0 + w * 16 + lp * 8 + r;
            int b = token >> 12, s = token & 4095;
            float v = acc[nt][r] + bval;
            size_t bh = (size_t)(b * NH + h);
            if (which == 2)       Vt[(bh * HD + d) * SEQ + s] = (__bf16)v;
            else if (which == 0)  Q[(bh * SEQ + s) * HD + d]  = (__bf16)v;
            else                  K[(bh * SEQ + s) * HD + d]  = (__bf16)v;
        }
    }
}

// ---------------- sliding window attention ----------------
// grid (nc*2=64, 16, 2), 128 threads (4 waves). dyn LDS: 64*384 f32
__global__ void k_win(const __bf16* __restrict__ Q, const __bf16* __restrict__ K,
                      const __bf16* __restrict__ Vt, float* __restrict__ ctx) {
    extern __shared__ float sc[];                 // [64][384]
    const int nsub = blockIdx.x, h = blockIdx.y, b = blockIdx.z;
    const int n = nsub >> 1, sub = nsub & 1;
    const int s0 = n * CH, qb = sub * 64;
    const int tid = threadIdx.x, w = tid >> 5, l = tid & 31, lp = l >> 4, ln = l & 15;
    const size_t bh = (size_t)(b * NH + h);
    const __bf16* Qh = Q  + bh * SEQ * HD;
    const __bf16* Kh = K  + bh * SEQ * HD;
    const __bf16* Vh = Vt + bh * HD * SEQ;

    // ---- scores: 16 rows x 384 keys per wave ----
    {
        const __bf16* arow = Qh + (size_t)(s0 + qb + w * 16 + ln) * HD;
        for (int nt = 0; nt < 24; ++nt) {
            int j = s0 - CH + nt * 16 + ln;
            j = j < 0 ? 0 : (j > SEQ - 1 ? SEQ - 1 : j);
            const __bf16* brow = Kh + (size_t)j * HD;
            v8f c = vzero();
            for (int kb = 0; kb < HD; kb += 32)
                c = wmma_bf16(load_a_frag(arow + kb, lp), load_b_frag(brow + kb, lp), c);
            for (int r = 0; r < 8; ++r)
                sc[(w * 16 + lp * 8 + r) * 384 + nt * 16 + ln] = c[r] * SCALE_F;
        }
    }
    __syncthreads();
    // ---- masked softmax, one thread per row ----
    if (tid < 64) {
        int qi = qb + tid;                        // chunk-relative query index
        float* row = sc + tid * 384;
        float mx = -1e30f;
        for (int m = 0; m < 384; ++m) {
            int ja = s0 - CH + m;
            bool ok = (m >= qi) && (m <= qi + 2 * CH) && (ja >= 0) && (ja < SEQ);
            float v = ok ? row[m] : -1e30f;
            row[m] = v;
            mx = fmaxf(mx, v);
        }
        float sum = 0.f;
        for (int m = 0; m < 384; ++m) { float p = __expf(row[m] - mx); sum += p; row[m] = p; }
        float inv = 1.f / sum;
        for (int m = 0; m < 384; ++m) row[m] *= inv;
    }
    __syncthreads();
    // ---- P @ V3 ----
    {
        v8f o[4]; for (int t = 0; t < 4; ++t) o[t] = vzero();
        for (int ms = 0; ms < 12; ++ms) {
            int kbm = ms * 32;
            const float* prow = sc + (w * 16 + ln) * 384 + kbm;
            v16bf a;
            for (int i = 0; i < 8; ++i) {
                a[i]     = (__bf16)prow[lp * 16 + i];
                a[8 + i] = (__bf16)prow[lp * 16 + 8 + i];
            }
            for (int nt = 0; nt < 4; ++nt) {
                int d = nt * 16 + ln;
                int j0 = s0 - CH + kbm + lp * 16;
                j0 = j0 < 0 ? 0 : (j0 > SEQ - 16 ? SEQ - 16 : j0);   // OOB rows have p==0
                v16bf bb = *(const v16bf*)(Vh + (size_t)d * SEQ + j0);
                o[nt] = wmma_bf16(a, bb, o[nt]);
            }
        }
        for (int nt = 0; nt < 4; ++nt)
            for (int r = 0; r < 8; ++r) {
                int srow = s0 + qb + w * 16 + lp * 8 + r;
                ctx[(bh * SEQ + srow) * HD + nt * 16 + ln] = o[nt][r];
            }
    }
}

// ---------------- global attention (two-pass flash) ----------------
__device__ inline void glob_scores(float* sc, const __bf16* Qh, const __bf16* Kh,
                                   int t, int w, int lp, int ln) {
    const __bf16* arow = Qh + (size_t)(w * 16 + ln) * HD;
    for (int nt = 0; nt < 8; ++nt) {
        int s = t * 128 + nt * 16 + ln;
        const __bf16* brow = Kh + (size_t)s * HD;
        v8f c = vzero();
        for (int kb = 0; kb < HD; kb += 32)
            c = wmma_bf16(load_a_frag(arow + kb, lp), load_b_frag(brow + kb, lp), c);
        for (int r = 0; r < 8; ++r)
            sc[(w * 16 + lp * 8 + r) * 128 + nt * 16 + ln] = c[r] * SCALE_F;
    }
}

// grid 32 (B*H), 128 threads. dyn LDS: 64*128 f32 + 128 f32 stats
__global__ void k_glob(const __bf16* __restrict__ Q, const __bf16* __restrict__ K,
                       const __bf16* __restrict__ Vt, const int* __restrict__ gnum,
                       float* __restrict__ ctx) {
    extern __shared__ float sm[];
    float* sc = sm;                 // [64][128]
    float* mrow = sm + 64 * 128;    // [64]
    float* lrow = mrow + 64;        // [64]
    const int G = *gnum;
    const int h = blockIdx.x & 15, b = blockIdx.x >> 4;
    const size_t bh = (size_t)(b * NH + h);
    const __bf16* Qh = Q  + bh * SEQ * HD;
    const __bf16* Kh = K  + bh * SEQ * HD;
    const __bf16* Vh = Vt + bh * HD * SEQ;
    const int tid = threadIdx.x, w = tid >> 5, l = tid & 31, lp = l >> 4, ln = l & 15;

    if (tid < 64) { mrow[tid] = -1e30f; lrow[tid] = 0.f; }
    __syncthreads();

    // pass 1: running max / sum over all 32 key tiles
    for (int t = 0; t < 32; ++t) {
        glob_scores(sc, Qh, Kh, t, w, lp, ln);
        __syncthreads();
        if (tid < 64) {
            float* row = sc + tid * 128;
            float tm = -1e30f;
            for (int m = 0; m < 128; ++m) tm = fmaxf(tm, row[m]);
            float nm = fmaxf(mrow[tid], tm);
            float ssum = 0.f;
            for (int m = 0; m < 128; ++m) ssum += __expf(row[m] - nm);
            lrow[tid] = lrow[tid] * __expf(mrow[tid] - nm) + ssum;
            mrow[tid] = nm;
        }
        __syncthreads();
    }
    // pass 2: recompute, normalize, accumulate O
    v8f o[4]; for (int t = 0; t < 4; ++t) o[t] = vzero();
    for (int t = 0; t < 32; ++t) {
        glob_scores(sc, Qh, Kh, t, w, lp, ln);
        __syncthreads();
        if (tid < 64) {
            float* row = sc + tid * 128;
            bool act = (tid < G) && (lrow[tid] > 0.f);
            float mr = mrow[tid], inv = act ? 1.f / lrow[tid] : 0.f;
            for (int m = 0; m < 128; ++m)
                row[m] = act ? __expf(row[m] - mr) * inv : 0.f;
        }
        __syncthreads();
        for (int ms = 0; ms < 4; ++ms) {
            const float* prow = sc + (w * 16 + ln) * 128 + ms * 32;
            v16bf a;
            for (int i = 0; i < 8; ++i) {
                a[i]     = (__bf16)prow[lp * 16 + i];
                a[8 + i] = (__bf16)prow[lp * 16 + 8 + i];
            }
            for (int nt = 0; nt < 4; ++nt) {
                int d = nt * 16 + ln;
                const __bf16* bp = Vh + (size_t)d * SEQ + t * 128 + ms * 32 + lp * 16;
                o[nt] = wmma_bf16(a, *(const v16bf*)bp, o[nt]);
            }
        }
        __syncthreads();
    }
    for (int nt = 0; nt < 4; ++nt)
        for (int r = 0; r < 8; ++r) {
            int g = w * 16 + lp * 8 + r;
            ctx[(bh * SEQ + g) * HD + nt * 16 + ln] += o[nt][r];
        }
}

// ---------------- output projection ----------------
// grid (64, 16), 256 threads
__global__ void k_oproj(const __bf16* __restrict__ ctxb, const __bf16* __restrict__ Wob,
                        const float* __restrict__ bo, float* __restrict__ out) {
    const int m0 = blockIdx.x * 128, n0 = blockIdx.y * 64;
    const int tid = threadIdx.x, w = tid >> 5, l = tid & 31, lp = l >> 4, ln = l & 15;
    const __bf16* arow = ctxb + (size_t)(m0 + w * 16 + ln) * EMB;

    v8f acc[4]; for (int t = 0; t < 4; ++t) acc[t] = vzero();
    for (int kb = 0; kb < EMB; kb += 32) {
        v16bf a = load_a_frag(arow + kb, lp);
        for (int nt = 0; nt < 4; ++nt) {
            int n = n0 + nt * 16 + ln;
            acc[nt] = wmma_bf16(a, load_b_frag(Wob + (size_t)n * EMB + kb, lp), acc[nt]);
        }
    }
    for (int nt = 0; nt < 4; ++nt) {
        int n = n0 + nt * 16 + ln;
        float bval = bo[n];
        for (int r = 0; r < 8; ++r) {
            int token = m0 + w * 16 + lp * 8 + r;
            out[(size_t)token * EMB + n] = acc[nt][r] + bval;
        }
    }
}

// ---------------- launcher ----------------
extern "C" void kernel_launch(void* const* d_in, const int* in_sizes, int n_in,
                              void* d_out, int out_size, void* d_ws, size_t ws_size,
                              hipStream_t stream) {
    const float* x  = (const float*)d_in[0];
    const float* Wq = (const float*)d_in[1];
    const float* bq = (const float*)d_in[2];
    const float* Wk = (const float*)d_in[3];
    const float* bk = (const float*)d_in[4];
    const float* Wv = (const float*)d_in[5];
    const float* bv = (const float*)d_in[6];
    const float* Wo = (const float*)d_in[7];
    const float* bo = (const float*)d_in[8];
    const int* gnum = (const int*)d_in[10];
    float* out = (float*)d_out;

    char* ws = (char*)d_ws;
    const size_t NTOK = (size_t)2 * SEQ;                 // 8192
    __bf16* xb   = (__bf16*)ws;                          ws += NTOK * EMB * 2;   // 16 MB (reused as ctxb)
    __bf16* Wqb  = (__bf16*)ws;                          ws += (size_t)EMB * EMB * 2;
    __bf16* Wkb  = (__bf16*)ws;                          ws += (size_t)EMB * EMB * 2;
    __bf16* Wvb  = (__bf16*)ws;                          ws += (size_t)EMB * EMB * 2;
    __bf16* Wob  = (__bf16*)ws;                          ws += (size_t)EMB * EMB * 2;
    __bf16* Qb   = (__bf16*)ws;                          ws += NTOK * EMB * 2;
    __bf16* Kb   = (__bf16*)ws;                          ws += NTOK * EMB * 2;
    __bf16* Vtb  = (__bf16*)ws;                          ws += NTOK * EMB * 2;
    float*  ctx  = (float*)ws;                           ws += NTOK * EMB * 4;

    k_cvt<<<32768, 256, 0, stream>>>(x,  xb,  (int)(NTOK * EMB));
    k_cvt<<<4096,  256, 0, stream>>>(Wq, Wqb, EMB * EMB);
    k_cvt<<<4096,  256, 0, stream>>>(Wk, Wkb, EMB * EMB);
    k_cvt<<<4096,  256, 0, stream>>>(Wv, Wvb, EMB * EMB);
    k_cvt<<<4096,  256, 0, stream>>>(Wo, Wob, EMB * EMB);

    k_qkv<<<dim3(64, 16, 3), 256, 0, stream>>>(xb, Wqb, Wkb, Wvb, bq, bk, bv, Qb, Kb, Vtb);

    k_win<<<dim3(64, 16, 2), 128, 64 * 384 * sizeof(float), stream>>>(Qb, Kb, Vtb, ctx);

    k_glob<<<32, 128, (64 * 128 + 128) * sizeof(float), stream>>>(Qb, Kb, Vtb, gnum, ctx);

    __bf16* ctxb = xb;                                   // alias: xb no longer needed
    k_cvt_ctx<<<32768, 256, 0, stream>>>(ctx, ctxb);

    k_oproj<<<dim3(64, 16), 256, 0, stream>>>(ctxb, Wob, bo, out);
}
